// ConversationalAttention_30511447671620
// MI455X (gfx1250) — matrix operand order repaired and checked by reading, not compiled, not run on hardware
//
#include <hip/hip_runtime.h>
#include <hip/hip_bf16.h>
#include <math.h>
#include <stdint.h>

// ---------------------------------------------------------------------------
// ConversationalAttention for MI455X (gfx1250, wave32, WMMA)
//   B=2, L=1024, D_MODEL=D_K=1024, H=16, HD=64
// Pipeline:
//   1) fp32 -> f16 downcast of activations + 6 weight matrices
//   2) 5 projections + final projection via V_WMMA_F32_16X16X32_F16:
//      B panel staged in LDS via global_load_async_to_lds_b128 (double buffer,
//      s_wait_asynccnt), A fragments register-double-buffered
//   3) fused RoPE + LayerNorm per (b,h,l) head vector -> f16 (B,H,L,HD)
//   4) scores = (QwKw^T + QsKs^T)/sqrt(128), causal mask -> raw logits in d_out
//   5) in-place row softmax over the attn tensor (1 read + 1 write of 128MB)
//   6) O = attn @ V via V_WMMA_F32_16X16X4_F32 on fp32 attn directly, V held
//      transposed (B,H,HD,L) so every fragment is one b64 load; causal K cut
// ---------------------------------------------------------------------------

typedef _Float16 half_t;
typedef __attribute__((ext_vector_type(16))) _Float16 v16h;
typedef __attribute__((ext_vector_type(8)))  _Float16 v8h;
typedef __attribute__((ext_vector_type(8)))  float    v8f;
typedef __attribute__((ext_vector_type(4)))  float    v4f;
typedef __attribute__((ext_vector_type(2)))  float    v2f;

#define B_  2
#define L_  1024
#define DM_ 1024
#define DK_ 1024
#define H_  16
#define HD_ 64

static __device__ __forceinline__ v16h cat8(v8h lo, v8h hi) {
  return __builtin_shufflevector(lo, hi, 0,1,2,3,4,5,6,7,8,9,10,11,12,13,14,15);
}

// -------------------------------------------------------- elementwise cast
__global__ void cvt_f32_to_f16(const float* __restrict__ src,
                               half_t* __restrict__ dst, int n) {
  int i = blockIdx.x * blockDim.x + threadIdx.x;
  if (i < n) dst[i] = (half_t)src[i];
}

// -------------------------------------------------------- V transpose
// pv (B,L,D_K) fp32 -> vt (B,H,HD,L) fp32
__global__ void v_transpose_kernel(const float* __restrict__ pv,
                                   float* __restrict__ vt) {
  const int idx = blockIdx.x * blockDim.x + threadIdx.x;   // < B*H*HD*L = 2M
  const int l = idx & (L_ - 1);
  const int d = (idx >> 10) & (HD_ - 1);
  const int h = (idx >> 16) & (H_ - 1);
  const int b = idx >> 20;
  vt[idx] = pv[(((long long)b << 10) + l) * DK_ + h * HD_ + d];
}

// -------------------------------------------------------- generic f16 GEMM
// C[m][n] = sum_k A[m][k] * Bm[n][k] + bias[n]   (f32 accumulate)
// grid = (N/64, M/128, batch); block = 256 (8 waves); wave -> 16 rows x 64 cols
// Shared 64x32-half B panel staged in LDS by async DMA, double-buffered.
__global__ void gemm_f16_wmma(const half_t* __restrict__ A,
                              const half_t* __restrict__ Bm,
                              const float*  __restrict__ bias,
                              float* __restrict__ C,
                              int K, int lda, int ldb, int ldc,
                              long long aBatch, long long bBatch, long long cBatch) {
  __shared__ half_t bTile[2][64 * 32];          // 2 x 4 KB
  const int t    = threadIdx.x;
  const int lane = t & 31;
  const int wave = t >> 5;
  const int hi   = lane >> 4;
  const int ln   = lane & 15;
  A  += (long long)blockIdx.z * aBatch;
  Bm += (long long)blockIdx.z * bBatch;
  C  += (long long)blockIdx.z * cBatch;
  const int m0 = blockIdx.y * 128 + wave * 16;
  const int n0 = blockIdx.x * 64;

  // async-copy mapping: thread t stages 16 bytes (8 halves) of the panel
  const int cpRow = t >> 2;                     // 0..63 -> column n0+cpRow
  const int cpCol = (t & 3) * 8;                // k-chunk within 32
  const half_t* gB = Bm + (long long)(n0 + cpRow) * ldb + cpCol;
  const unsigned ldsBase = (unsigned)(uintptr_t)(&bTile[0][0]);
  const unsigned ldsDst  = ldsBase + (unsigned)t * 16u;

  // stage k=0 panel into buffer 0
  asm volatile("global_load_async_to_lds_b128 %0, %1, off"
               :: "v"(ldsDst), "v"(gB) : "memory");
  asm volatile("s_wait_asynccnt 0" ::: "memory");
  __syncthreads();

  const half_t* aRow = A + (long long)(m0 + ln) * lda;
  const int c1 = hi * 8;                        // A K-chunk offset (ISA layout)
  v16h aCur = cat8(*(const v8h*)(aRow + c1), *(const v8h*)(aRow + c1 + 16));

  v8f acc[4] = {};
  for (int k = 0; k < K; k += 32) {
    const int buf = (k >> 5) & 1;
    const bool more = (k + 32 < K);
    // prefetch next A fragment (registers; clamp keeps addresses valid)
    const int kn = more ? k + 32 : k;
    v8h nlo = *(const v8h*)(aRow + kn + c1);
    v8h nhi = *(const v8h*)(aRow + kn + c1 + 16);

    // read B fragments for this k-step from LDS
    v16h bfr[4];
#pragma unroll
    for (int j = 0; j < 4; ++j) {
      const half_t* bp = &bTile[buf][(j * 16 + ln) * 32 + hi * 8];
      bfr[j] = cat8(*(const v8h*)bp, *(const v8h*)(bp + 16));
    }
    __syncthreads();                            // whole WG done reading bTile[buf]

    if (more) {                                 // DMA next panel while we compute
      const half_t* gBk = gB + k + 32;
      const unsigned dst = ldsBase + (unsigned)((buf ^ 1) << 12) + (unsigned)t * 16u;
      asm volatile("global_load_async_to_lds_b128 %0, %1, off"
                   :: "v"(dst), "v"(gBk) : "memory");
    }
#pragma unroll
    for (int j = 0; j < 4; ++j)
      acc[j] = __builtin_amdgcn_wmma_f32_16x16x32_f16(
          false, aCur, false, bfr[j], (short)0, acc[j], false, false);

    aCur = cat8(nlo, nhi);
    asm volatile("s_wait_asynccnt 0" ::: "memory");
    __syncthreads();                            // next panel visible WG-wide
  }

#pragma unroll
  for (int j = 0; j < 4; ++j) {
    const int col = n0 + j * 16 + ln;
    const float bv = bias ? bias[col] : 0.0f;
#pragma unroll
    for (int r = 0; r < 8; ++r) {
      const int row = m0 + r + hi * 8;
      C[(long long)row * ldc + col] = acc[j][r] + bv;
    }
  }
}

// -------------------------------------------------------- fused RoPE + LN
// grid = (L, B*H, 4 arrays); block = 64 (one head vector)
__global__ void rope_ln_kernel(const float* __restrict__ pqw, const float* __restrict__ pkw,
                               const float* __restrict__ pqs, const float* __restrict__ pks,
                               half_t* __restrict__ oqw, half_t* __restrict__ okw,
                               half_t* __restrict__ oqs, half_t* __restrict__ oks,
                               const float* __restrict__ gamma,
                               const float* __restrict__ beta) {
  const int l  = blockIdx.x;
  const int bh = blockIdx.y;                 // b*H + h
  const int b  = bh / H_, h = bh % H_;
  const int d  = threadIdx.x;                // 0..63

  const float* src; half_t* dst;
  switch (blockIdx.z) {
    case 0:  src = pqw; dst = oqw; break;
    case 1:  src = pkw; dst = okw; break;
    case 2:  src = pqs; dst = oqs; break;
    default: src = pks; dst = oks; break;
  }
  const float* x = src + ((long long)b * L_ + l) * DK_ + h * HD_;

  const int i = d & 31;                      // frequency index
  const float ang = (float)l * expf((float)i * (-logf(10000.0f) / 32.0f));
  const float sn = sinf(ang), cs = cosf(ang);
  const float x1 = x[2*i], x2 = x[2*i + 1];
  const float r = (d < 32) ? (x1*cs - x2*sn) : (x1*sn + x2*cs);

  __shared__ float red[4];
  float s = r, q = r * r;
  for (int off = 16; off; off >>= 1) {
    s += __shfl_xor(s, off, 32);
    q += __shfl_xor(q, off, 32);
  }
  const int wv = d >> 5;
  if ((d & 31) == 0) { red[wv] = s; red[2 + wv] = q; }
  __syncthreads();
  const float mu  = (red[0] + red[1]) * (1.0f / 64.0f);
  const float var = (red[2] + red[3]) * (1.0f / 64.0f) - mu * mu;
  const float y = (r - mu) * rsqrtf(var + 1e-5f) * gamma[d] + beta[d];
  dst[((long long)bh * L_ + l) * HD_ + d] = (half_t)y;
}

// -------------------------------------------------------- causal score kernel
// grid = (L/64, L/128, B*H); block = 256; writes raw masked logits
__global__ void scores_kernel(const half_t* __restrict__ qwh, const half_t* __restrict__ kwh,
                              const half_t* __restrict__ qsh, const half_t* __restrict__ ksh,
                              float* __restrict__ attn) {
  const int lane = threadIdx.x & 31;
  const int wave = threadIdx.x >> 5;
  const int hi = lane >> 4, ln = lane & 15;
  const int bh = blockIdx.z;
  const int m0 = blockIdx.y * 128 + wave * 16;
  const int n0 = blockIdx.x * 64;
  float* S = attn + (long long)bh * L_ * L_;

  if (n0 > m0 + 15) {                        // wave tile fully above diagonal
#pragma unroll
    for (int j = 0; j < 4; ++j)
#pragma unroll
      for (int r = 0; r < 8; ++r)
        S[(long long)(m0 + r + hi*8) * L_ + (n0 + j*16 + ln)] = -1e30f;
    return;
  }

  const long long base = (long long)bh * L_ * HD_;
  const half_t* qwRow = qwh + base + (long long)(m0 + ln) * HD_;
  const half_t* qsRow = qsh + base + (long long)(m0 + ln) * HD_;
  const int c1 = hi * 16;

  v16h aqw[2], aqs[2];
#pragma unroll
  for (int t = 0; t < 2; ++t) {
    const int k0 = t * 32;
    aqw[t] = cat8(*(const v8h*)(qwRow + k0 + c1), *(const v8h*)(qwRow + k0 + c1 + 8));
    aqs[t] = cat8(*(const v8h*)(qsRow + k0 + c1), *(const v8h*)(qsRow + k0 + c1 + 8));
  }

  v8f acc[4] = {};
#pragma unroll
  for (int j = 0; j < 4; ++j) {
    const half_t* kwRow = kwh + base + (long long)(n0 + j*16 + ln) * HD_ + hi*16;
    const half_t* ksRow = ksh + base + (long long)(n0 + j*16 + ln) * HD_ + hi*16;
#pragma unroll
    for (int t = 0; t < 2; ++t) {
      const int k0 = t * 32;
      v16h bkw = cat8(*(const v8h*)(kwRow + k0), *(const v8h*)(kwRow + k0 + 8));
      acc[j] = __builtin_amdgcn_wmma_f32_16x16x32_f16(
          false, aqw[t], false, bkw, (short)0, acc[j], false, false);
      v16h bks = cat8(*(const v8h*)(ksRow + k0), *(const v8h*)(ksRow + k0 + 8));
      acc[j] = __builtin_amdgcn_wmma_f32_16x16x32_f16(
          false, aqs[t], false, bks, (short)0, acc[j], false, false);
    }
  }

  const float sc = 0.08838834764831845f;     // 1/sqrt(2*HD)
#pragma unroll
  for (int j = 0; j < 4; ++j) {
    const int col = n0 + j*16 + ln;
#pragma unroll
    for (int r = 0; r < 8; ++r) {
      const int row = m0 + r + hi*8;
      S[(long long)row * L_ + col] = (col <= row) ? acc[j][r] * sc : -1e30f;
    }
  }
}

// -------------------------------------------------------- in-place row softmax
// grid = B*H*L; block = 128 (4 waves); each thread owns 8 contiguous elems
__global__ void softmax_kernel(float* __restrict__ attn) {
  const int t = threadIdx.x;
  float* p = attn + (long long)blockIdx.x * L_ + t * 8;
  v4f a = *(const v4f*)p;
  v4f b = *(const v4f*)(p + 4);

  float mx = fmaxf(fmaxf(fmaxf(a[0], a[1]), fmaxf(a[2], a[3])),
                   fmaxf(fmaxf(b[0], b[1]), fmaxf(b[2], b[3])));
  for (int off = 16; off; off >>= 1) mx = fmaxf(mx, __shfl_xor(mx, off, 32));
  __shared__ float red[8];
  const int wv = t >> 5;
  if ((t & 31) == 0) red[wv] = mx;
  __syncthreads();
  mx = fmaxf(fmaxf(red[0], red[1]), fmaxf(red[2], red[3]));

  float e[8]; float sum = 0.0f;
#pragma unroll
  for (int i = 0; i < 4; ++i) { e[i]     = expf(a[i] - mx); sum += e[i]; }
#pragma unroll
  for (int i = 0; i < 4; ++i) { e[4 + i] = expf(b[i] - mx); sum += e[4 + i]; }
  for (int off = 16; off; off >>= 1) sum += __shfl_xor(sum, off, 32);
  if ((t & 31) == 0) red[4 + wv] = sum;
  __syncthreads();
  const float inv = 1.0f / (red[4] + red[5] + red[6] + red[7]);

  v4f o1 = { e[0]*inv, e[1]*inv, e[2]*inv, e[3]*inv };
  v4f o2 = { e[4]*inv, e[5]*inv, e[6]*inv, e[7]*inv };
  *(v4f*)p = o1;
  *(v4f*)(p + 4) = o2;
}

// -------------------------------------------------------- O = attn @ V (fp32 WMMA)
// grid = (1, L/128, B*H); block = 256; causal K truncation; output (B,L,D_K)
// vt is (B,H,HD,L) so every B fragment is one contiguous b64 load.
__global__ void attn_v_kernel(const float* __restrict__ attn,
                              const float* __restrict__ vt,
                              float* __restrict__ O) {
  const int lane = threadIdx.x & 31;
  const int wave = threadIdx.x >> 5;
  const int hi = lane >> 4, ln = lane & 15;
  const int bh = blockIdx.z;
  const int b = bh / H_, h = bh % H_;
  const int m0 = blockIdx.y * 128 + wave * 16;

  const float* A = attn + (long long)bh * L_ * L_ + (long long)(m0 + ln) * L_ + hi * 2;
  const float* Vt[4];
#pragma unroll
  for (int j = 0; j < 4; ++j)
    Vt[j] = vt + ((long long)bh * HD_ + j * 16 + ln) * L_ + hi * 2;

  v8f acc[4] = {};
  const int kEnd = m0 + 16;                  // attn == 0 beyond the diagonal

  v2f aCur = *(const v2f*)(A);
  v2f bCur[4];
#pragma unroll
  for (int j = 0; j < 4; ++j) bCur[j] = *(const v2f*)(Vt[j]);

  for (int k = 0; k < kEnd; k += 4) {
    const int kn = (k + 4 < kEnd) ? k + 4 : k;   // clamped prefetch
    v2f aN = *(const v2f*)(A + kn);
    v2f bN[4];
#pragma unroll
    for (int j = 0; j < 4; ++j) bN[j] = *(const v2f*)(Vt[j] + kn);
#pragma unroll
    for (int j = 0; j < 4; ++j)
      acc[j] = __builtin_amdgcn_wmma_f32_16x16x4_f32(
          false, aCur, false, bCur[j], (short)0, acc[j], false, false);
    aCur = aN;
#pragma unroll
    for (int j = 0; j < 4; ++j) bCur[j] = bN[j];
  }
#pragma unroll
  for (int j = 0; j < 4; ++j) {
    const int col = h * HD_ + j * 16 + ln;
#pragma unroll
    for (int r = 0; r < 8; ++r) {
      const int row = m0 + r + hi * 8;
      O[((long long)b * L_ + row) * DK_ + col] = acc[j][r];
    }
  }
}

// ---------------------------------------------------------------------------
extern "C" void kernel_launch(void* const* d_in, const int* in_sizes, int n_in,
                              void* d_out, int out_size, void* d_ws, size_t ws_size,
                              hipStream_t stream) {
  (void)in_sizes; (void)n_in; (void)out_size; (void)ws_size;
  const float* word  = (const float*)d_in[0];
  const float* spk   = (const float*)d_in[1];
  /* d_in[2] = tril mask: causal structure handled analytically */
  const float* qw_w  = (const float*)d_in[3];
  const float* qw_b  = (const float*)d_in[4];
  const float* kw_w  = (const float*)d_in[5];
  const float* kw_b  = (const float*)d_in[6];
  const float* qs_w  = (const float*)d_in[7];
  const float* qs_b  = (const float*)d_in[8];
  const float* ks_w  = (const float*)d_in[9];
  const float* ks_b  = (const float*)d_in[10];
  const float* v_w   = (const float*)d_in[11];
  const float* v_b   = (const float*)d_in[12];
  const float* out_w = (const float*)d_in[13];
  const float* out_b = (const float*)d_in[14];
  const float* gamma = (const float*)d_in[15];
  const float* beta  = (const float*)d_in[16];

  const size_t nX = (size_t)B_ * L_ * DM_;   // 2M activations per tensor
  const size_t nW = (size_t)DK_ * DM_;       // 1M per weight matrix

  char* ws = (char*)d_ws;
  size_t off = 0;
  auto alloc = [&](size_t bytes) -> char* {
    char* p = ws + off;
    off += (bytes + 255) & ~(size_t)255;
    return p;
  };
  half_t* xw_h  = (half_t*)alloc(nX * 2);
  half_t* xs_h  = (half_t*)alloc(nX * 2);
  half_t* wq_h  = (half_t*)alloc(nW * 2);
  half_t* wk_h  = (half_t*)alloc(nW * 2);
  half_t* wqs_h = (half_t*)alloc(nW * 2);
  half_t* wks_h = (half_t*)alloc(nW * 2);
  half_t* wv_h  = (half_t*)alloc(nW * 2);
  half_t* wo_h  = (half_t*)alloc(nW * 2);
  float*  pqw   = (float*)alloc(nX * 4);
  float*  pkw   = (float*)alloc(nX * 4);
  float*  pqs   = (float*)alloc(nX * 4);
  float*  pks   = (float*)alloc(nX * 4);
  float*  pv    = (float*)alloc(nX * 4);
  float*  vt    = (float*)alloc(nX * 4);     // (B,H,HD,L)
  half_t* qwh   = (half_t*)alloc(nX * 2);    // (B,H,L,HD)
  half_t* kwh   = (half_t*)alloc(nX * 2);
  half_t* qsh   = (half_t*)alloc(nX * 2);
  half_t* ksh   = (half_t*)alloc(nX * 2);
  float*  O32   = (float*)alloc(nX * 4);     // (B,L,D_K)
  half_t* Oh    = (half_t*)alloc(nX * 2);

  float* outp = (float*)d_out;
  float* attn = outp + nX;                   // (B,H,L,L) region of d_out

  // 1) downcasts
  const int cbX = (int)((nX + 255) / 256);
  const int cbW = (int)((nW + 255) / 256);
  cvt_f32_to_f16<<<cbX, 256, 0, stream>>>(word,  xw_h, (int)nX);
  cvt_f32_to_f16<<<cbX, 256, 0, stream>>>(spk,   xs_h, (int)nX);
  cvt_f32_to_f16<<<cbW, 256, 0, stream>>>(qw_w,  wq_h,  (int)nW);
  cvt_f32_to_f16<<<cbW, 256, 0, stream>>>(kw_w,  wk_h,  (int)nW);
  cvt_f32_to_f16<<<cbW, 256, 0, stream>>>(qs_w,  wqs_h, (int)nW);
  cvt_f32_to_f16<<<cbW, 256, 0, stream>>>(ks_w,  wks_h, (int)nW);
  cvt_f32_to_f16<<<cbW, 256, 0, stream>>>(v_w,   wv_h,  (int)nW);
  cvt_f32_to_f16<<<cbW, 256, 0, stream>>>(out_w, wo_h,  (int)nW);

  // 2) five projections (M=2048, N=1024, K=1024)
  dim3 gg(DK_ / 64, (B_ * L_) / 128, 1);
  gemm_f16_wmma<<<gg, 256, 0, stream>>>(xw_h, wq_h,  qw_b, pqw, DM_, DM_, DM_, DK_, 0, 0, 0);
  gemm_f16_wmma<<<gg, 256, 0, stream>>>(xw_h, wk_h,  kw_b, pkw, DM_, DM_, DM_, DK_, 0, 0, 0);
  gemm_f16_wmma<<<gg, 256, 0, stream>>>(xs_h, wqs_h, qs_b, pqs, DM_, DM_, DM_, DK_, 0, 0, 0);
  gemm_f16_wmma<<<gg, 256, 0, stream>>>(xs_h, wks_h, ks_b, pks, DM_, DM_, DM_, DK_, 0, 0, 0);
  gemm_f16_wmma<<<gg, 256, 0, stream>>>(xw_h, wv_h,  v_b,  pv,  DM_, DM_, DM_, DK_, 0, 0, 0);

  // 3) RoPE + LayerNorm -> f16 head tensors; V transpose for attn_v
  rope_ln_kernel<<<dim3(L_, B_ * H_, 4), 64, 0, stream>>>(
      pqw, pkw, pqs, pks, qwh, kwh, qsh, ksh, gamma, beta);
  v_transpose_kernel<<<cbX, 256, 0, stream>>>(pv, vt);

  // 4) raw causal logits into d_out attn region
  scores_kernel<<<dim3(L_ / 64, L_ / 128, B_ * H_), 256, 0, stream>>>(
      qwh, kwh, qsh, ksh, attn);

  // 5) in-place softmax
  softmax_kernel<<<dim3(B_ * H_ * L_), 128, 0, stream>>>(attn);

  // 6) O = attn @ V (fp32 WMMA, causal truncation)
  attn_v_kernel<<<dim3(1, L_ / 128, B_ * H_), 256, 0, stream>>>(attn, vt, O32);

  // 7) final projection
  cvt_f32_to_f16<<<cbX, 256, 0, stream>>>(O32, Oh, (int)nX);
  gemm_f16_wmma<<<gg, 256, 0, stream>>>(Oh, wo_h, out_b, outp, DK_, DK_, DK_, DM_, 0, 0, 0);
}
